// TreeDecoder_46007689675208
// MI455X (gfx1250) — compile-verified, matrix operand order checked by reading, not run on hardware
//
#include <hip/hip_runtime.h>

// ---------------------------------------------------------------------------
// TreeDecoder for MI455X (gfx1250, wave32, WMMA + async-LDS DMA).
// Pipeline:
//   f32->f16 weight/activation conversion (conv weights re-laid k-major)
//   4x [GEMM(f16 WMMA, f32 acc) -> BN column stats -> apply BN+LeakyReLU -> f16]
//   3x [async-DMA gather-to-LDS -> per-sample WMMA GEMM + sum/sumsq
//       -> TreeLayerNorm + LeakyReLU]
// ---------------------------------------------------------------------------

typedef __attribute__((ext_vector_type(16))) _Float16 v16h;
typedef __attribute__((ext_vector_type(2)))  _Float16 v2h;
typedef __attribute__((ext_vector_type(8)))  float    v8f;

#define LRELU(v) ((v) > 0.f ? (v) : 0.01f * (v))

// Load a 16-lane-row x 32(K) f16 fragment (A layout per ISA 7.12.2: lane holds
// row; VGPR v pairs cover K = (v/4)*16 + half*8 + (v%4)*2). The B fragment uses
// the mirrored layout (lane = column, VGPRs over K), so the same loader reads
// row `n` of the row-major (N x K) weight matrix.
__device__ __forceinline__ v16h load_frag_k(const _Float16* rowPtrAtK0, int half) {
  v16h r;
#pragma unroll
  for (int v = 0; v < 8; ++v) {
    const int kk = ((v >> 2) << 4) + (half << 3) + ((v & 3) << 1);
    v2h t = *(const v2h*)(rowPtrAtK0 + kk);
    r[2 * v]     = t.x;
    r[2 * v + 1] = t.y;
  }
  return r;
}

// ---------------------------------------------------------------------------
__global__ void cvt_kernel(const float* __restrict__ src,
                           _Float16* __restrict__ dst, int n) {
  int i = blockIdx.x * blockDim.x + threadIdx.x;
  if (i < n) dst[i] = (_Float16)src[i];
}

// Conv weights (O,C,3) f32 -> k-major f16: dst[o][k*C+c] = src[o][c][k].
__global__ void cvt_convw_kernel(const float* __restrict__ src,
                                 _Float16* __restrict__ dst, int O, int C) {
  int i = blockIdx.x * blockDim.x + threadIdx.x;
  if (i >= O * C * 3) return;
  int o = i / (3 * C);
  int r = i - o * 3 * C;
  int k = r / C;
  int c = r - k * C;
  dst[i] = (_Float16)src[((size_t)o * C + c) * 3 + k];
}

// ---------------------------------------------------------------------------
// Y[M=2048, N] = X[M, K](f16) @ W[N, K](f16)^T + bias.  Block = 128x64 tile,
// 8 waves as 4x2, each wave owns 2x2 WMMA 16x16 tiles, K stepped by 32.
__global__ __launch_bounds__(256) void gemm_f16_kernel(
    const _Float16* __restrict__ X, const _Float16* __restrict__ W,
    const float* __restrict__ bias, float* __restrict__ Y, int N, int K) {
  const int tid  = threadIdx.x;
  const int lane = tid & 31;
  const int wave = tid >> 5;
  const int half = lane >> 4;
  const int ln16 = lane & 15;
  const int wm   = wave >> 1;                 // 0..3
  const int wn   = wave & 1;                  // 0..1
  const int row0 = blockIdx.y * 128 + wm * 32;
  const int col0 = blockIdx.x * 64  + wn * 32;

  v8f acc[2][2] = {};
  for (int k0 = 0; k0 < K; k0 += 32) {
    v16h a[2], b[2];
#pragma unroll
    for (int mt = 0; mt < 2; ++mt)
      a[mt] = load_frag_k(X + (size_t)(row0 + mt * 16 + ln16) * K + k0, half);
#pragma unroll
    for (int nt = 0; nt < 2; ++nt)
      b[nt] = load_frag_k(W + (size_t)(col0 + nt * 16 + ln16) * K + k0, half);
#pragma unroll
    for (int mt = 0; mt < 2; ++mt)
#pragma unroll
      for (int nt = 0; nt < 2; ++nt)
        acc[mt][nt] = __builtin_amdgcn_wmma_f32_16x16x32_f16(
            false, a[mt], false, b[nt], (short)0, acc[mt][nt], false, false);
  }
#pragma unroll
  for (int mt = 0; mt < 2; ++mt)
#pragma unroll
    for (int nt = 0; nt < 2; ++nt) {
      const int col = col0 + nt * 16 + ln16;
      const float bv = bias[col];
#pragma unroll
      for (int r = 0; r < 8; ++r) {
        const int row = row0 + mt * 16 + r + half * 8;  // C/D layout
        Y[(size_t)row * N + col] = acc[mt][nt][r] + bv;
      }
    }
}

// ---------------------------------------------------------------------------
// Per-feature batch stats over M=2048 rows -> scale/shift for fused BN.
__global__ __launch_bounds__(256) void bn_stats_kernel(
    const float* __restrict__ Y, const float* __restrict__ g,
    const float* __restrict__ be, float* __restrict__ scale,
    float* __restrict__ shift, int N) {
  const int n = blockIdx.x;
  const int tid = threadIdx.x;
  float s = 0.f, q = 0.f;
  for (int r = tid; r < 2048; r += 256) {
    float v = Y[(size_t)r * N + n];
    s += v; q += v * v;
  }
  for (int off = 16; off > 0; off >>= 1) {
    s += __shfl_xor(s, off);
    q += __shfl_xor(q, off);
  }
  __shared__ float ss[8], sq[8];
  if ((tid & 31) == 0) { ss[tid >> 5] = s; sq[tid >> 5] = q; }
  __syncthreads();
  if (tid == 0) {
    float S = 0.f, Q = 0.f;
    for (int i = 0; i < 8; ++i) { S += ss[i]; Q += sq[i]; }
    const float invM = 1.0f / 2048.0f;
    const float mean = S * invM;
    const float var  = fmaxf(Q * invM - mean * mean, 0.f);  // biased, as jnp.var
    const float sc = g[n] * rsqrtf(var + 1e-5f);
    scale[n] = sc;
    shift[n] = be[n] - mean * sc;
  }
}

// BN + LeakyReLU, emit f16 activations for the next GEMM.
__global__ void bn_apply_kernel(const float* __restrict__ Y,
                                const float* __restrict__ scale,
                                const float* __restrict__ shift,
                                _Float16* __restrict__ Xn, int total, int N) {
  int i = blockIdx.x * blockDim.x + threadIdx.x;
  if (i >= total) return;
  int n = i % N;
  float v = Y[i] * scale[n] + shift[n];
  Xn[i] = (_Float16)LRELU(v);
}

// Layer-4 variant: also transpose (B,4096)->(B, node=64, ch=64) for the convs.
__global__ void bn_apply_tr_kernel(const float* __restrict__ Y,
                                   const float* __restrict__ scale,
                                   const float* __restrict__ shift,
                                   _Float16* __restrict__ Xc) {
  int i = blockIdx.x * blockDim.x + threadIdx.x;
  if (i >= 2048 * 4096) return;
  int b = i >> 12, f = i & 4095;
  int c = f >> 6, n = f & 63;
  float v = Y[i] * scale[f] + shift[f];
  Xc[((size_t)(b * 64 + n)) * 64 + c] = (_Float16)LRELU(v);
}

// ---------------------------------------------------------------------------
// One workgroup per sample. The expanded matrix is k-major: E[m][k*C+c] =
// x[b][idx[3m+k]][c], so each gathered triple-entry is a contiguous 2C-byte
// row copy -> per-lane 16B GLOBAL_LOAD_ASYNC_TO_LDS_B128 DMA (ASYNCcnt).
// Then WMMA GEMM against k-major (O x 3C) weights, writing pre-norm
// Y[b][o][1+m] and reducing sum/sumsq for TreeLayerNorm.
template <int C, int O>
__global__ __launch_bounds__(256) void tree_conv_kernel(
    const _Float16* __restrict__ Xin,   // [B][64 nodes][C] f16
    const int* __restrict__ idx,        // [B][189]
    const _Float16* __restrict__ Wc,    // [O][3C] f16, k-major
    const float* __restrict__ bias,     // [O]
    float* __restrict__ Y,              // [B][O][64] pre-norm (node 0 skipped)
    float* __restrict__ stats) {        // [B][2] mean, std(ddof=1)
  constexpr int Kc  = 3 * C;
  constexpr int LDE = Kc + 16;          // (3C+16)*2 bytes per row: 16B multiple
  constexpr int CH  = C / 8;            // 16-byte chunks per gathered row
  extern __shared__ __align__(16) _Float16 E[];  // [64][LDE]
  __shared__ int   sIdx[192];
  __shared__ float sSum[8], sSq[8];

  const int b    = blockIdx.x;
  const int tid  = threadIdx.x;
  const int lane = tid & 31, wave = tid >> 5;
  const int half = lane >> 4, ln16 = lane & 15;

  for (int t = tid; t < 189; t += 256) sIdx[t] = idx[b * 189 + t];
  for (int j = tid; j < LDE; j += 256) E[63 * LDE + j] = (_Float16)0.f;  // pad row
  // Warm the cache with this wave's first weight rows while the DMA runs.
  if (wave < O / 16)
    __builtin_prefetch(Wc + (size_t)(wave * 16 + ln16) * Kc, 0, 0);
  __syncthreads();                      // sIdx visible to all waves

  // Async gather: global [node row] -> LDS E row segment, 16B per lane.
  const unsigned ebase = (unsigned)(unsigned long long)&E[0];
  const _Float16* xb = Xin + (size_t)b * 64 * C;
  for (int e = tid; e < 189 * CH; e += 256) {
    const int t = e / CH;               // CH is a power of two
    const int q = e - t * CH;
    const int m = t / 3;
    const int k = t - 3 * m;
    const unsigned ldst = ebase + (unsigned)((m * LDE + k * C + q * 8) * 2);
    const unsigned long long gsrc =
        (unsigned long long)(xb + (size_t)sIdx[t] * C + q * 8);
    asm volatile("global_load_async_to_lds_b128 %0, %1, off"
                 :: "v"(ldst), "v"(gsrc) : "memory");
  }
  asm volatile("s_wait_asynccnt 0x0" ::: "memory");
  __syncthreads();

  float lsum = 0.f, lsq = 0.f;
  for (int nt = wave; nt < O / 16; nt += 8) {
    const int col = nt * 16 + ln16;
    const _Float16* wrow = Wc + (size_t)col * Kc;
    v8f acc[4] = {};
    for (int k0 = 0; k0 < Kc; k0 += 32) {
      const v16h bfr = load_frag_k(wrow + k0, half);   // 1 B-frag ...
#pragma unroll
      for (int mt = 0; mt < 4; ++mt) {                  // ... 4 WMMAs
        v16h a = load_frag_k(&E[(mt * 16 + ln16) * LDE + k0], half);
        acc[mt] = __builtin_amdgcn_wmma_f32_16x16x32_f16(
            false, a, false, bfr, (short)0, acc[mt], false, false);
      }
    }
    const float bv = bias[col];
#pragma unroll
    for (int mt = 0; mt < 4; ++mt)
#pragma unroll
      for (int r = 0; r < 8; ++r) {
        const int row = mt * 16 + r + half * 8;
        if (row < 63) {                                 // row 63 is zero-padding
          const float v = acc[mt][r] + bv;
          Y[((size_t)b * O + col) * 64 + 1 + row] = v;
          lsum += v; lsq += v * v;
        }
      }
  }
  for (int off = 16; off > 0; off >>= 1) {
    lsum += __shfl_xor(lsum, off);
    lsq  += __shfl_xor(lsq, off);
  }
  if (lane == 0) { sSum[wave] = lsum; sSq[wave] = lsq; }
  __syncthreads();
  if (tid == 0) {
    float S = 0.f, Q = 0.f;
    for (int i = 0; i < 8; ++i) { S += sSum[i]; Q += sSq[i]; }
    const float cnt  = (float)(O * 64);          // includes zero null-node col
    const float mean = S / cnt;
    const float var  = fmaxf((Q - S * S / cnt) / (cnt - 1.0f), 0.f);  // ddof=1
    stats[2 * b]     = mean;
    stats[2 * b + 1] = sqrtf(var);
  }
}

// TreeLayerNorm + LeakyReLU -> f16 [b][node][c] for the next conv's gather.
__global__ void conv_apply_kernel(const float* __restrict__ Y,
                                  const float* __restrict__ stats,
                                  _Float16* __restrict__ Xn, int O) {
  int i = blockIdx.x * blockDim.x + threadIdx.x;
  if (i >= 2048 * O * 64) return;
  int node = i & 63;
  int o    = (i >> 6) % O;
  int b    = i / (O * 64);
  float v  = node ? Y[i] : 0.f;                 // null node is zero pre-norm
  v = (v - stats[2 * b]) / (stats[2 * b + 1] + 1e-5f);
  Xn[((size_t)(b * 64 + node)) * O + o] = (_Float16)LRELU(v);
}

// Final conv: normalize in place in f32 (d_out already holds pre-norm [b][o][n]).
__global__ void conv_apply_final_kernel(float* __restrict__ Y,
                                        const float* __restrict__ stats, int O) {
  int i = blockIdx.x * blockDim.x + threadIdx.x;
  if (i >= 2048 * O * 64) return;
  int node = i & 63;
  int b    = i / (O * 64);
  float v  = node ? Y[i] : 0.f;
  v = (v - stats[2 * b]) / (stats[2 * b + 1] + 1e-5f);
  Y[i] = LRELU(v);
}

// ---------------------------------------------------------------------------
extern "C" void kernel_launch(void* const* d_in, const int* in_sizes, int n_in,
                              void* d_out, int out_size, void* d_ws, size_t ws_size,
                              hipStream_t stream) {
  (void)in_sizes; (void)n_in; (void)out_size; (void)ws_size;
  const float* trees = (const float*)d_in[0];
  const int*   idx   = (const int*)d_in[1];
  const float *w1 = (const float*)d_in[2],  *b1 = (const float*)d_in[3];
  const float *g1 = (const float*)d_in[4],  *be1 = (const float*)d_in[5];
  const float *w2 = (const float*)d_in[6],  *b2 = (const float*)d_in[7];
  const float *g2 = (const float*)d_in[8],  *be2 = (const float*)d_in[9];
  const float *w3 = (const float*)d_in[10], *b3 = (const float*)d_in[11];
  const float *g3 = (const float*)d_in[12], *be3 = (const float*)d_in[13];
  const float *w4 = (const float*)d_in[14], *b4 = (const float*)d_in[15];
  const float *g4 = (const float*)d_in[16], *be4 = (const float*)d_in[17];
  const float *cw1 = (const float*)d_in[18], *cb1 = (const float*)d_in[19];
  const float *cw2 = (const float*)d_in[20], *cb2 = (const float*)d_in[21];
  const float *cw3 = (const float*)d_in[22], *cb3 = (const float*)d_in[23];

  // Workspace bump allocator (256B aligned).
  char* p = (char*)d_ws;
  auto alloc = [&](size_t bytes) -> void* {
    char* r = p;
    p += (bytes + 255) & ~(size_t)255;
    return (void*)r;
  };
  _Float16* wf1  = (_Float16*)alloc((size_t)64 * 64 * 2);
  _Float16* wf2  = (_Float16*)alloc((size_t)256 * 64 * 2);
  _Float16* wf3  = (_Float16*)alloc((size_t)1024 * 256 * 2);
  _Float16* wf4  = (_Float16*)alloc((size_t)4096 * 1024 * 2);
  _Float16* wfc1 = (_Float16*)alloc((size_t)128 * 192 * 2);
  _Float16* wfc2 = (_Float16*)alloc((size_t)256 * 384 * 2);
  _Float16* wfc3 = (_Float16*)alloc((size_t)512 * 768 * 2);
  _Float16* Xa   = (_Float16*)alloc((size_t)2048 * 64 * 256 * 2);  // ping (67MB)
  _Float16* Xb   = (_Float16*)alloc((size_t)2048 * 64 * 128 * 2);  // pong (34MB)
  float*    Ym   = (float*)alloc((size_t)2048 * 4096 * 4);         // MLP pre-BN
  float*    Yc   = (float*)alloc((size_t)2048 * 256 * 64 * 4);     // conv1/2 pre-norm
  float*    scl  = (float*)alloc((size_t)4096 * 4);
  float*    sft  = (float*)alloc((size_t)4096 * 4);
  float*    stats = (float*)alloc((size_t)2048 * 2 * 4);

  auto cvt = [&](const float* s, _Float16* d, int n) {
    cvt_kernel<<<(n + 255) / 256, 256, 0, stream>>>(s, d, n);
  };
  auto cvtw = [&](const float* s, _Float16* d, int O, int C) {
    cvt_convw_kernel<<<(O * C * 3 + 255) / 256, 256, 0, stream>>>(s, d, O, C);
  };
  cvt(w1, wf1, 64 * 64);       cvt(w2, wf2, 256 * 64);
  cvt(w3, wf3, 1024 * 256);    cvt(w4, wf4, 4096 * 1024);
  cvtw(cw1, wfc1, 128, 64);    cvtw(cw2, wfc2, 256, 128);
  cvtw(cw3, wfc3, 512, 256);
  cvt(trees, Xa, 2048 * 64);   // X0

  auto gemm = [&](const _Float16* X, const _Float16* W, const float* bias,
                  float* Y, int N, int K) {
    dim3 grid(N / 64, 2048 / 128);
    gemm_f16_kernel<<<grid, 256, 0, stream>>>(X, W, bias, Y, N, K);
  };
  auto bn = [&](float* Y, const float* g, const float* be, int N) {
    bn_stats_kernel<<<N, 256, 0, stream>>>(Y, g, be, scl, sft, N);
  };

  // ---- MLP ----
  gemm(Xa, wf1, b1, Ym, 64, 64);
  bn(Ym, g1, be1, 64);
  bn_apply_kernel<<<(2048 * 64 + 255) / 256, 256, 0, stream>>>(Ym, scl, sft, Xb, 2048 * 64, 64);

  gemm(Xb, wf2, b2, Ym, 256, 64);
  bn(Ym, g2, be2, 256);
  bn_apply_kernel<<<(2048 * 256 + 255) / 256, 256, 0, stream>>>(Ym, scl, sft, Xa, 2048 * 256, 256);

  gemm(Xa, wf3, b3, Ym, 1024, 256);
  bn(Ym, g3, be3, 1024);
  bn_apply_kernel<<<(2048 * 1024 + 255) / 256, 256, 0, stream>>>(Ym, scl, sft, Xb, 2048 * 1024, 1024);

  gemm(Xb, wf4, b4, Ym, 4096, 1024);
  bn(Ym, g4, be4, 4096);
  bn_apply_tr_kernel<<<(2048 * 4096 + 255) / 256, 256, 0, stream>>>(Ym, scl, sft, Xa);

  // ---- Tree convolutions ----
  tree_conv_kernel<64, 128><<<2048, 256, (size_t)(3 * 64 + 16) * 64 * 2, stream>>>(
      Xa, idx, wfc1, cb1, Yc, stats);
  conv_apply_kernel<<<(2048 * 128 * 64 + 255) / 256, 256, 0, stream>>>(Yc, stats, Xb, 128);

  tree_conv_kernel<128, 256><<<2048, 256, (size_t)(3 * 128 + 16) * 64 * 2, stream>>>(
      Xb, idx, wfc2, cb2, Yc, stats);
  conv_apply_kernel<<<(2048 * 256 * 64 + 255) / 256, 256, 0, stream>>>(Yc, stats, Xa, 256);

  tree_conv_kernel<256, 512><<<2048, 256, (size_t)(3 * 256 + 16) * 64 * 2, stream>>>(
      Xa, idx, wfc3, cb3, (float*)d_out, stats);
  conv_apply_final_kernel<<<(2048 * 512 * 64 + 255) / 256, 256, 0, stream>>>(
      (float*)d_out, stats, 512);
}
